// BrainGT_68856915689662
// MI455X (gfx1250) — matrix operand order repaired and checked by reading, not compiled, not run on hardware
//
#include <hip/hip_runtime.h>

// ---------------------------------------------------------------------------
// Problem constants (from reference): L=2048, D=1024, H=16, hd=64, HID=4096
// ---------------------------------------------------------------------------
#define LL   2048
#define DM   1024
#define NH   16
#define HDIM 64
#define HID  4096
#define EPSF 1e-5f

typedef __bf16 bf16;
typedef __attribute__((ext_vector_type(16))) __bf16 v16bf;
typedef __attribute__((ext_vector_type(8)))  __bf16 v8bf;
typedef __attribute__((ext_vector_type(8)))  float  v8f;

// ---------------------------------------------------------------------------
// WMMA helpers (CDNA5 16x16x32 bf16, fp32 accum)
// A-frag (16x32, MxK): lane m=lane&15; vec[0..15] = A[m][16*half+0..15]
// B-frag (32x16, KxN): lane n=lane&15; vec[0..15] = B[16*half+0..15][n]
// C/D  (16x16 f32):    elem r <-> (row r+8*half, col lane&15)
// ---------------------------------------------------------------------------
__device__ __forceinline__ v8f wmma_bf16(v16bf a, v16bf b, v8f c) {
  return __builtin_amdgcn_wmma_f32_16x16x32_bf16(
      /*neg_a=*/false, a, /*neg_b=*/false, b,
      /*c_mod=*/(short)0, c, /*reuse_a=*/false, /*reuse_b=*/false);
}

__device__ __forceinline__ v16bf load_afrag(const bf16* __restrict__ row, int half) {
  v8bf lo = *reinterpret_cast<const v8bf*>(row + 16 * half);
  v8bf hi = *reinterpret_cast<const v8bf*>(row + 16 * half + 8);
  v16bf a;
#pragma unroll
  for (int i = 0; i < 8; ++i) { a[i] = lo[i]; a[i + 8] = hi[i]; }
  return a;
}

__device__ __forceinline__ v16bf load_afrag_lds(const bf16* row, int half) {
  v16bf a;
#pragma unroll
  for (int i = 0; i < 8; ++i) { a[i] = row[16 * half + i]; a[i + 8] = row[16 * half + 8 + i]; }
  return a;
}

// ---------------------------------------------------------------------------
// Generic bf16 GEMM:  out[m][n] = sum_k A[m][k] * W[n][k] + bias[n]
// A: [M x K] bf16 row-major, W: [N x K] bf16 row-major (torch Linear weight).
// One wave computes a 32x64 tile (8 accumulators): two A-fragments share the
// four B-fragments per k-step -> 8 WMMAs / 192 B-per-lane (~21 FLOP/B from
// L2) and 8 independent accumulator chains for hazard hiding.
// Optional ReLU; optional fp32 / bf16 / transposed-bf16 outputs.
// ---------------------------------------------------------------------------
__global__ void gemm_bf16_kernel(const bf16* __restrict__ A, const bf16* __restrict__ W,
                                 const float* __restrict__ bias,
                                 float* __restrict__ outF, bf16* __restrict__ outB,
                                 bf16* __restrict__ outBT,
                                 int M, int N, int K, int doRelu) {
  const int wid    = (int)((blockIdx.x * blockDim.x + threadIdx.x) >> 5);
  const int lane   = (int)(threadIdx.x & 31);
  const int tilesN = N >> 6;
  const int total  = (M >> 5) * tilesN;
  if (wid >= total) return;
  const int tm = wid / tilesN, tn = wid % tilesN;
  const int mbase = tm << 5, nbase = tn << 6;
  const int half = lane >> 4, l15 = lane & 15;

  const bf16* arow0 = A + (size_t)(mbase +      l15) * K;
  const bf16* arow1 = A + (size_t)(mbase + 16 + l15) * K;
  const bf16* wr0   = W + (size_t)(nbase +  0 + l15) * K;
  const bf16* wr1   = W + (size_t)(nbase + 16 + l15) * K;
  const bf16* wr2   = W + (size_t)(nbase + 32 + l15) * K;
  const bf16* wr3   = W + (size_t)(nbase + 48 + l15) * K;

  v8f c[2][4] = {{{}, {}, {}, {}}, {{}, {}, {}, {}}};
  for (int kb = 0; kb < K; kb += 32) {
    v16bf a0 = load_afrag(arow0 + kb, half);
    v16bf a1 = load_afrag(arow1 + kb, half);
    v16bf b0 = *reinterpret_cast<const v16bf*>(wr0 + kb + 16 * half);
    v16bf b1 = *reinterpret_cast<const v16bf*>(wr1 + kb + 16 * half);
    v16bf b2 = *reinterpret_cast<const v16bf*>(wr2 + kb + 16 * half);
    v16bf b3 = *reinterpret_cast<const v16bf*>(wr3 + kb + 16 * half);
    c[0][0] = wmma_bf16(a0, b0, c[0][0]);
    c[1][0] = wmma_bf16(a1, b0, c[1][0]);
    c[0][1] = wmma_bf16(a0, b1, c[0][1]);
    c[1][1] = wmma_bf16(a1, b1, c[1][1]);
    c[0][2] = wmma_bf16(a0, b2, c[0][2]);
    c[1][2] = wmma_bf16(a1, b2, c[1][2]);
    c[0][3] = wmma_bf16(a0, b3, c[0][3]);
    c[1][3] = wmma_bf16(a1, b3, c[1][3]);
  }

  float bb[4];
#pragma unroll
  for (int t = 0; t < 4; ++t) bb[t] = bias[nbase + 16 * t + l15];

#pragma unroll
  for (int i = 0; i < 2; ++i) {
#pragma unroll
    for (int r = 0; r < 8; ++r) {
      const int row = mbase + 16 * i + r + 8 * half;
      const size_t rowoff = (size_t)row * N;
#pragma unroll
      for (int t = 0; t < 4; ++t) {
        float v = c[i][t][r] + bb[t];
        if (doRelu) v = fmaxf(v, 0.f);
        const int col = nbase + 16 * t + l15;
        if (outF)  outF[rowoff + col] = v;
        if (outB)  outB[rowoff + col] = (bf16)v;
        if (outBT) outBT[(size_t)col * M + row] = (bf16)v;  // direct vT layout
      }
    }
  }
}

// ---------------------------------------------------------------------------
// Flash attention, one wave per (head, 16-query-row tile).
// S = Q·K^T * 0.125 + sp_bias ; online softmax over 2048 keys in blocks of 32;
// P staged via per-wave LDS tile (C-layout -> A-layout); O += P·V via WMMA.
// q,k: [L x 1024] bf16 row-major; vT: [1024 x L] bf16 (vT[c][n] = v[n][c]);
// spb: [L x L] f32 softmax(shortest_path)*0.5; out attn: [L x 1024] bf16.
// ---------------------------------------------------------------------------
__global__ void attn_kernel(const bf16* __restrict__ q, const bf16* __restrict__ k,
                            const bf16* __restrict__ vT, const float* __restrict__ spb,
                            bf16* __restrict__ attn) {
  __shared__ bf16 Plds[8][16 * 32];  // per-wave 16x32 P tile (1 KB each)

  const int w    = (int)(threadIdx.x >> 5);
  const int wid  = (int)(blockIdx.x * (blockDim.x >> 5) + w);
  if (wid >= NH * (LL / 16)) return;
  const int h    = wid >> 7;          // head
  const int qt   = wid & 127;         // query tile
  const int lane = (int)(threadIdx.x & 31);
  const int half = lane >> 4, l15 = lane & 15;
  const int mbase = qt << 4;

  const bf16* qrow = q + (size_t)(mbase + l15) * DM + h * HDIM;
  v16bf aQ0 = load_afrag(qrow, half);        // head-dim k = 0..31
  v16bf aQ1 = load_afrag(qrow + 32, half);   // head-dim k = 32..63

  v8f o[4] = {{}, {}, {}, {}};
  float rmax[8], rsum[8];
#pragma unroll
  for (int r = 0; r < 8; ++r) { rmax[r] = -1e30f; rsum[r] = 0.f; }

  bf16* pl = &Plds[w][0];

  for (int kb = 0; kb < LL; kb += 32) {
    v8f s[2];
#pragma unroll
    for (int j = 0; j < 2; ++j) {
      const bf16* krow = k + (size_t)(kb + 16 * j + l15) * DM + h * HDIM;
      v16bf b0 = *reinterpret_cast<const v16bf*>(krow + 16 * half);
      v16bf b1 = *reinterpret_cast<const v16bf*>(krow + 32 + 16 * half);
      v8f sc = {};
      sc = wmma_bf16(aQ0, b0, sc);
      sc = wmma_bf16(aQ1, b1, sc);
#pragma unroll
      for (int r = 0; r < 8; ++r) {
        float bias = spb[(size_t)(mbase + r + 8 * half) * LL + kb + 16 * j + l15];
        sc[r] = sc[r] * 0.125f + bias;
      }
      s[j] = sc;
    }

    // online softmax update (rows live across 16-lane halves; xor masks 1..8
    // stay within each half of the wave32)
#pragma unroll
    for (int r = 0; r < 8; ++r) {
      float t = fmaxf(s[0][r], s[1][r]);
      t = fmaxf(t, __shfl_xor(t, 1));
      t = fmaxf(t, __shfl_xor(t, 2));
      t = fmaxf(t, __shfl_xor(t, 4));
      t = fmaxf(t, __shfl_xor(t, 8));
      const float mnew = fmaxf(rmax[r], t);
      const float corr = __expf(rmax[r] - mnew);
      rmax[r] = mnew;
      const float p0 = __expf(s[0][r] - mnew);
      const float p1 = __expf(s[1][r] - mnew);
      s[0][r] = p0; s[1][r] = p1;
      float ps = p0 + p1;
      ps += __shfl_xor(ps, 1);
      ps += __shfl_xor(ps, 2);
      ps += __shfl_xor(ps, 4);
      ps += __shfl_xor(ps, 8);
      rsum[r] = rsum[r] * corr + ps;
      o[0][r] *= corr; o[1][r] *= corr; o[2][r] *= corr; o[3][r] *= corr;
    }

    // C-layout -> LDS (row-major 16x32) -> A-layout re-read (same wave, DS in-order)
#pragma unroll
    for (int r = 0; r < 8; ++r) {
      pl[(r + 8 * half) * 32 + l15]      = (bf16)s[0][r];
      pl[(r + 8 * half) * 32 + 16 + l15] = (bf16)s[1][r];
    }
    v16bf aP = load_afrag_lds(pl + l15 * 32, half);

#pragma unroll
    for (int t = 0; t < 4; ++t) {
      const bf16* vrow = vT + (size_t)(h * HDIM + 16 * t + l15) * LL + kb + 16 * half;
      v16bf bv = *reinterpret_cast<const v16bf*>(vrow);
      o[t] = wmma_bf16(aP, bv, o[t]);
    }
  }

#pragma unroll
  for (int r = 0; r < 8; ++r) {
    const float inv = 1.f / rsum[r];
    const size_t rowoff = (size_t)(mbase + r + 8 * half) * DM + h * HDIM;
#pragma unroll
    for (int t = 0; t < 4; ++t)
      attn[rowoff + 16 * t + l15] = (bf16)(o[t][r] * inv);
  }
}

// ---------------------------------------------------------------------------
// softmax(shortest_path, axis=-1) * 0.5, one block (256 thr) per row of 2048
// ---------------------------------------------------------------------------
__global__ void sp_softmax_kernel(const float* __restrict__ sp, float* __restrict__ out) {
  const int row = blockIdx.x;
  const int tid = (int)threadIdx.x;
  const float* r = sp + (size_t)row * LL;
  __shared__ float red[256];

  float m = -1e30f;
  for (int c = tid; c < LL; c += 256) m = fmaxf(m, r[c]);
  red[tid] = m; __syncthreads();
  for (int s = 128; s > 0; s >>= 1) { if (tid < s) red[tid] = fmaxf(red[tid], red[tid + s]); __syncthreads(); }
  m = red[0]; __syncthreads();

  float sum = 0.f;
  for (int c = tid; c < LL; c += 256) sum += __expf(r[c] - m);
  red[tid] = sum; __syncthreads();
  for (int s = 128; s > 0; s >>= 1) { if (tid < s) red[tid] += red[tid + s]; __syncthreads(); }
  const float inv = 0.5f / red[0];

  float* o = out + (size_t)row * LL;
  for (int c = tid; c < LL; c += 256) o[c] = __expf(r[c] - m) * inv;
}

// ---------------------------------------------------------------------------
// h = LayerNorm(a + b) * g + be ; one block per row (D=1024, 256 thr x 4)
// optional fp32 and bf16 outputs
// ---------------------------------------------------------------------------
__global__ void add_ln_kernel(const float* __restrict__ a, const float* __restrict__ b,
                              const float* __restrict__ g, const float* __restrict__ be,
                              float* __restrict__ outF, bf16* __restrict__ outB) {
  const int row = blockIdx.x;
  const int tid = (int)threadIdx.x;
  __shared__ float red[256];
  const float* ar = a + (size_t)row * DM;
  const float* br = b + (size_t)row * DM;

  float vals[4];
  float s = 0.f;
#pragma unroll
  for (int i = 0; i < 4; ++i) { vals[i] = ar[tid + 256 * i] + br[tid + 256 * i]; s += vals[i]; }
  red[tid] = s; __syncthreads();
  for (int st = 128; st > 0; st >>= 1) { if (tid < st) red[tid] += red[tid + st]; __syncthreads(); }
  const float mu = red[0] * (1.f / DM); __syncthreads();

  float s2 = 0.f;
#pragma unroll
  for (int i = 0; i < 4; ++i) { const float d = vals[i] - mu; s2 += d * d; }
  red[tid] = s2; __syncthreads();
  for (int st = 128; st > 0; st >>= 1) { if (tid < st) red[tid] += red[tid + st]; __syncthreads(); }
  const float inv = rsqrtf(red[0] * (1.f / DM) + EPSF);

#pragma unroll
  for (int i = 0; i < 4; ++i) {
    const int c = tid + 256 * i;
    const float v = (vals[i] - mu) * inv * g[c] + be[c];
    if (outF) outF[(size_t)row * DM + c] = v;
    if (outB) outB[(size_t)row * DM + c] = (bf16)v;
  }
}

// ---------------------------------------------------------------------------
// fp32 -> bf16 convert
// ---------------------------------------------------------------------------
__global__ void cvt_kernel(const float* __restrict__ s, bf16* __restrict__ d, int n) {
  for (int i = (int)(blockIdx.x * blockDim.x + threadIdx.x); i < n;
       i += (int)(gridDim.x * blockDim.x))
    d[i] = (bf16)s[i];
}

// ---------------------------------------------------------------------------
// Host-side orchestration
// ---------------------------------------------------------------------------
extern "C" void kernel_launch(void* const* d_in, const int* in_sizes, int n_in,
                              void* d_out, int out_size, void* d_ws, size_t ws_size,
                              hipStream_t stream) {
  const float* x   = (const float*)d_in[0];
  const float* sp  = (const float*)d_in[1];
  const float* Wq  = (const float*)d_in[2];  const float* bq  = (const float*)d_in[3];
  const float* Wk  = (const float*)d_in[4];  const float* bk  = (const float*)d_in[5];
  const float* Wv  = (const float*)d_in[6];  const float* bv  = (const float*)d_in[7];
  const float* Wqp = (const float*)d_in[8];  const float* bqp = (const float*)d_in[9];
  const float* Wkp = (const float*)d_in[10]; const float* bkp = (const float*)d_in[11];
  const float* Wvp = (const float*)d_in[12]; const float* bvp = (const float*)d_in[13];
  const float* Wo  = (const float*)d_in[14]; const float* bo  = (const float*)d_in[15];
  const float* W1  = (const float*)d_in[16]; const float* b1  = (const float*)d_in[17];
  const float* W2  = (const float*)d_in[18]; const float* b2  = (const float*)d_in[19];
  const float* g1  = (const float*)d_in[20]; const float* be1 = (const float*)d_in[21];
  const float* g2  = (const float*)d_in[22]; const float* be2 = (const float*)d_in[23];
  float* out = (float*)d_out;

  char* ws = (char*)d_ws;
  size_t off = 0;
  auto alloc = [&](size_t bytes) -> void* {
    void* p = ws + off;
    off = (off + bytes + 255) & ~(size_t)255;
    return p;
  };

  bf16* xb   = (bf16*)alloc((size_t)LL * DM * 2);
  bf16* wqb  = (bf16*)alloc((size_t)DM * DM * 2);
  bf16* wkb  = (bf16*)alloc((size_t)DM * DM * 2);
  bf16* wvb  = (bf16*)alloc((size_t)DM * DM * 2);
  bf16* wqpb = (bf16*)alloc((size_t)DM * DM * 2);
  bf16* wkpb = (bf16*)alloc((size_t)DM * DM * 2);
  bf16* wvpb = (bf16*)alloc((size_t)DM * DM * 2);
  bf16* wob  = (bf16*)alloc((size_t)DM * DM * 2);
  bf16* w1b  = (bf16*)alloc((size_t)HID * DM * 2);
  bf16* w2b  = (bf16*)alloc((size_t)DM * HID * 2);
  float* spb = (float*)alloc((size_t)LL * LL * 4);
  bf16* tmp1 = (bf16*)alloc((size_t)LL * DM * 2);
  bf16* qb   = (bf16*)alloc((size_t)LL * DM * 2);
  bf16* kb_  = (bf16*)alloc((size_t)LL * DM * 2);
  bf16* vT   = (bf16*)alloc((size_t)DM * LL * 2);
  bf16* atnb = (bf16*)alloc((size_t)LL * DM * 2);
  float* of  = (float*)alloc((size_t)LL * DM * 4);
  float* h1f = (float*)alloc((size_t)LL * DM * 4);
  bf16* h1b  = (bf16*)alloc((size_t)LL * DM * 2);
  bf16* f1b  = (bf16*)alloc((size_t)LL * HID * 2);
  float* f2f = (float*)alloc((size_t)LL * DM * 4);
  (void)ws_size; (void)n_in; (void)in_sizes; (void)out_size;

  auto cvt = [&](const float* s, bf16* d, int n) {
    cvt_kernel<<<1024, 256, 0, stream>>>(s, d, n);
  };
  auto gemm = [&](const bf16* A, const bf16* Wt, const float* bias,
                  float* oF, bf16* oB, bf16* oBT, int M, int N, int K, int relu) {
    const int waves  = (M / 32) * (N / 64);
    const int blocks = (waves + 7) / 8;  // 8 waves per 256-thread block
    gemm_bf16_kernel<<<blocks, 256, 0, stream>>>(A, Wt, bias, oF, oB, oBT, M, N, K, relu);
  };

  // Stage 0: precision downcast of activations and weights (bf16 WMMA path)
  cvt(x,   xb,   LL * DM);
  cvt(Wq,  wqb,  DM * DM);  cvt(Wk,  wkb,  DM * DM);  cvt(Wv,  wvb,  DM * DM);
  cvt(Wqp, wqpb, DM * DM);  cvt(Wkp, wkpb, DM * DM);  cvt(Wvp, wvpb, DM * DM);
  cvt(Wo,  wob,  DM * DM);
  cvt(W1,  w1b,  HID * DM); cvt(W2,  w2b,  DM * HID);

  // Stage 1: spatial bias = softmax(shortest_path) * 0.5
  sp_softmax_kernel<<<LL, 256, 0, stream>>>(sp, spb);

  // Stage 2: q/k/v double projections (2048x1024x1024 bf16 GEMMs);
  //          the v-chain writes the transposed [1024 x 2048] layout directly.
  gemm(xb,   wqb,  bq,  nullptr, tmp1, nullptr, LL, DM, DM, 0);
  gemm(tmp1, wqpb, bqp, nullptr, qb,   nullptr, LL, DM, DM, 0);
  gemm(xb,   wkb,  bk,  nullptr, tmp1, nullptr, LL, DM, DM, 0);
  gemm(tmp1, wkpb, bkp, nullptr, kb_,  nullptr, LL, DM, DM, 0);
  gemm(xb,   wvb,  bv,  nullptr, tmp1, nullptr, LL, DM, DM, 0);
  gemm(tmp1, wvpb, bvp, nullptr, nullptr, vT,   LL, DM, DM, 0);

  // Stage 3: flash attention (2048 waves = 16 heads x 128 q-tiles; 8 waves/block)
  attn_kernel<<<(NH * (LL / 16)) / 8, 256, 0, stream>>>(qb, kb_, vT, spb, atnb);

  // Stage 4: output projection + residual + LN1
  gemm(atnb, wob, bo, of, nullptr, nullptr, LL, DM, DM, 0);
  add_ln_kernel<<<LL, 256, 0, stream>>>(x, of, g1, be1, h1f, h1b);

  // Stage 5: FFN with fused ReLU, then residual + LN2 into d_out
  gemm(h1b, w1b, b1, nullptr, f1b, nullptr, LL, HID, DM, 1);
  gemm(f1b, w2b, b2, f2f, nullptr, nullptr, LL, DM, HID, 1);
  add_ln_kernel<<<LL, 256, 0, stream>>>(h1f, f2f, g2, be2, out, nullptr);
}
